// DeformableCompositeTransformerDecoderLayer_39453569581672
// MI455X (gfx1250) — matrix operand order repaired and checked by reading, not compiled, not run on hardware
//
#include <hip/hip_runtime.h>

// ---------------------------------------------------------------------------
// DeformableCompositeTransformerDecoderLayer for MI455X (gfx1250, wave32).
// GEMMs: bf16 WMMA (v_wmma_f32_16x16x32_bf16, f32 accum), 128x64 block tile,
// software-pipelined staging: b128 global loads for slab k+1 issued while
// WMMAs consume slab k from LDS (packed bf16, b64 stores, b128 frag reads).
// Attention / LN / deformable sampling are wave32 micro-kernels (4 waves per
// block) using cross-lane shuffles.
// ---------------------------------------------------------------------------

#define D_MODEL  256
#define N_HEADS  8
#define DH       32
#define D_FFN    1024
#define BATCH    8
#define N_OBJ    100
#define N_PTS    16
#define N_TEXT   25
#define S_TOT    19947   // 100*150 + 50*75 + 25*38 + 13*19

typedef __attribute__((ext_vector_type(16))) __bf16 v16bf;
typedef __attribute__((ext_vector_type(8)))  float  v8f;

union FragB16 { v16bf v; unsigned int u[8]; };
union FragF32 { v8f v; float f[8]; };

__device__ __forceinline__ unsigned int pack2bf(float lo, float hi) {
  unsigned int ul = __builtin_bit_cast(unsigned int, lo);
  unsigned int uh = __builtin_bit_cast(unsigned int, hi);
  ul += 0x7FFFu + ((ul >> 16) & 1u);        // round-to-nearest-even
  uh += 0x7FFFu + ((uh >> 16) & 1u);
  return (ul >> 16) | (uh & 0xFFFF0000u);
}

// ---------------------------------------------------------------------------
// GEMM: C[M,N] = A[M,K] @ W[N,K]^T + bias[N]  (optional fused ReLU)
// Block tile 128x64, 8 waves, K-step 32. Each wave: one 16-row A fragment,
// four 16-col B fragments -> 4 WMMAs per K-step.
// OOB M rows are clamped (their D rows are computed but never stored), so
// staging is branchless. Global loads for the next K-slab are issued right
// after the barrier and retire while the current slab's WMMAs execute.
// ---------------------------------------------------------------------------
template <int RELU>
__global__ void __launch_bounds__(256)
gemm_bf16_wmma(const float* __restrict__ A, const float* __restrict__ W,
               const float* __restrict__ bias, float* __restrict__ C,
               int M, int N, int K)
{
  __shared__ unsigned int Alds[128 * 16];   // 128 rows x 32 k as bf16 pairs
  __shared__ unsigned int Blds[64 * 16];    // 64 cols  x 32 k as bf16 pairs

  const int tid  = threadIdx.x;
  const int wave = tid >> 5;
  const int lane = tid & 31;
  const int half = lane >> 4;               // K-half selector (k = 16*half + s)
  const int lx   = lane & 15;
  const int m0   = blockIdx.x * 128;
  const int n0   = blockIdx.y * 64;

  // K-invariant staging coordinates (row clamp makes staging branchless)
  size_t aOff[4];                           // &A[row*K + c4*4]
#pragma unroll
  for (int i = 0; i < 4; ++i) {
    int c = tid + i * 256;                  // 1024 float4 chunks of A tile
    int mm = c >> 3, c4 = c & 7;
    int row = m0 + mm; row = (row < M) ? row : (M - 1);
    aOff[i] = (size_t)row * K + c4 * 4;
  }
  size_t bOff[2];
#pragma unroll
  for (int i = 0; i < 2; ++i) {
    int c = tid + i * 256;                  // 512 float4 chunks of B tile
    bOff[i] = (size_t)(n0 + (c >> 3)) * K + (c & 7) * 4;
  }

  FragF32 acc[4];
#pragma unroll
  for (int j = 0; j < 4; ++j)
#pragma unroll
    for (int r = 0; r < 8; ++r) acc[j].f[r] = 0.f;

  // prologue: slab 0 loads into registers
  float4 fa[4], fb[2];
#pragma unroll
  for (int i = 0; i < 4; ++i) fa[i] = *(const float4*)&A[aOff[i]];
#pragma unroll
  for (int i = 0; i < 2; ++i) fb[i] = *(const float4*)&W[bOff[i]];

  for (int k0 = 0; k0 < K; k0 += 32) {
    // --- pack current registers into LDS (bf16 pairs, b64 stores) ---
#pragma unroll
    for (int i = 0; i < 4; ++i) {
      int c = tid + i * 256;
      int mm = c >> 3, c4 = c & 7;
      uint2 pk; pk.x = pack2bf(fa[i].x, fa[i].y); pk.y = pack2bf(fa[i].z, fa[i].w);
      *(uint2*)&Alds[mm * 16 + c4 * 2] = pk;
    }
#pragma unroll
    for (int i = 0; i < 2; ++i) {
      int c = tid + i * 256;
      int nn = c >> 3, c4 = c & 7;
      uint2 pk; pk.x = pack2bf(fb[i].x, fb[i].y); pk.y = pack2bf(fb[i].z, fb[i].w);
      *(uint2*)&Blds[nn * 16 + c4 * 2] = pk;
    }
    __syncthreads();

    // --- issue next-slab global loads; they retire during the WMMAs ---
    const int k1 = k0 + 32;
    if (k1 < K) {
#pragma unroll
      for (int i = 0; i < 4; ++i) fa[i] = *(const float4*)&A[aOff[i] + k1];
#pragma unroll
      for (int i = 0; i < 2; ++i) fb[i] = *(const float4*)&W[bOff[i] + k1];
    }

    // --- fragments in contiguous k order (k = 16*half + s), same for A and B ---
    FragB16 a, b[4];
    const int am = wave * 16 + lx;
#pragma unroll
    for (int p = 0; p < 8; ++p) {           // A: k = 2p + half*16
      int ku = p + half * 8;
      a.u[p] = Alds[am * 16 + ku];
    }
#pragma unroll
    for (int j = 0; j < 4; ++j)
#pragma unroll
      for (int p = 0; p < 8; ++p)           // B: n = lane&15, k = 2p + half*16
        b[j].u[p] = Blds[(j * 16 + lx) * 16 + p + half * 8];

#pragma unroll
    for (int j = 0; j < 4; ++j)
      acc[j].v = __builtin_amdgcn_wmma_f32_16x16x32_bf16(
          false, a.v, false, b[j].v, (short)0, acc[j].v, false, false);
    __syncthreads();
  }

  // --- epilogue: D layout = VGPR r -> M = r + half*8, N = lane&15 ---
#pragma unroll
  for (int j = 0; j < 4; ++j) {
    int col = n0 + j * 16 + lx;
    float bv = bias[col];
#pragma unroll
    for (int r = 0; r < 8; ++r) {
      int row = m0 + wave * 16 + half * 8 + r;
      if (row >= M) continue;
      float o = acc[j].f[r] + bv;
      if (RELU) o = fmaxf(o, 0.f);
      C[(size_t)row * N + col] = o;
    }
  }
}

// ---------------------------------------------------------------------------
// Small-sequence attention: 4 waves/block, 1 wave per (query, head, sequence).
// dh == 32 == wave32 lanes; online softmax with cross-lane dot product.
// mode 0: intra (seq over points,  stride 1,    L = nPts, seq = b*nObj+o)
// mode 1: inter (seq over objects, stride nPts, L = nObj, seq = b*nPts+pt)
// ---------------------------------------------------------------------------
__global__ void __launch_bounds__(128)
flash_attn_small(const float* __restrict__ Q, const float* __restrict__ K,
                 const float* __restrict__ V, float* __restrict__ O,
                 int mode, int nObj, int nPts)
{
  const int wave = threadIdx.x >> 5, lane = threadIdx.x & 31;
  const int h = blockIdx.y, s = blockIdx.z;
  int base, stride, L;
  if (mode == 0) { base = s * nPts; stride = 1; L = nPts; }
  else { int b = s / nPts, pt = s - b * nPts; base = b * nObj * nPts + pt; stride = nPts; L = nObj; }
  const int qi = blockIdx.x * 4 + wave;
  if (qi >= L) return;

  const size_t qoff = (size_t)(base + qi * stride) * D_MODEL + h * DH + lane;
  const float qv = Q[qoff];
  float mrun = -1e30f, lrun = 0.f, acc = 0.f;
  for (int kk = 0; kk < L; ++kk) {
    size_t toff = (size_t)(base + kk * stride) * D_MODEL + h * DH + lane;
    float part = qv * K[toff];
#pragma unroll
    for (int d = 16; d; d >>= 1) part += __shfl_xor(part, d, 32);
    float sc   = part * 0.17677669529663689f;   // 1/sqrt(32)
    float mn   = fmaxf(mrun, sc);
    float corr = __expf(mrun - mn);
    float p    = __expf(sc - mn);
    lrun = lrun * corr + p;
    acc  = acc  * corr + p * V[toff];
    mrun = mn;
  }
  O[qoff] = acc / lrun;
}

// ---------------------------------------------------------------------------
// LayerNorm(X (+R)) * g + b : 4 waves/block, one wave per 256-wide row.
// ---------------------------------------------------------------------------
__global__ void __launch_bounds__(128)
ln_residual(const float* __restrict__ X, const float* __restrict__ R,
            const float* __restrict__ g, const float* __restrict__ b,
            float* __restrict__ Y, int rows)
{
  const int wave = threadIdx.x >> 5, lane = threadIdx.x & 31;
  const int row = blockIdx.x * 4 + wave;
  if (row >= rows) return;
  float x[8]; float s = 0.f;
#pragma unroll
  for (int i = 0; i < 8; ++i) {
    int d = lane + i * 32;
    float v = X[(size_t)row * D_MODEL + d];
    if (R) v += R[(size_t)row * D_MODEL + d];
    x[i] = v; s += v;
  }
#pragma unroll
  for (int d = 16; d; d >>= 1) s += __shfl_xor(s, d, 32);
  float mean = s * (1.f / 256.f);
  float vs = 0.f;
#pragma unroll
  for (int i = 0; i < 8; ++i) { float t = x[i] - mean; vs += t * t; }
#pragma unroll
  for (int d = 16; d; d >>= 1) vs += __shfl_xor(vs, d, 32);
  float inv = rsqrtf(vs * (1.f / 256.f) + 1e-5f);
#pragma unroll
  for (int i = 0; i < 8; ++i) {
    int d = lane + i * 32;
    Y[(size_t)row * D_MODEL + d] = (x[i] - mean) * inv * g[d] + b[d];
  }
}

__global__ void __launch_bounds__(256)
add_vec4(const float4* __restrict__ a, const float4* __restrict__ b,
         float4* __restrict__ o, int n4)
{
  int i = blockIdx.x * blockDim.x + threadIdx.x;
  if (i < n4) {
    float4 x = a[i], y = b[i];
    float4 r; r.x = x.x + y.x; r.y = x.y + y.y; r.z = x.z + y.z; r.w = x.w + y.w;
    o[i] = r;
  }
}

// ---------------------------------------------------------------------------
// Multi-scale deformable attention sampling. 4 waves/block, one wave per
// (b, query, head); lane = channel. 16-logit softmax + lane-broadcast offsets
// + bilinear gather (grid_sample align_corners=False, zero padding).
// Spatial shapes are static per the reference -> hardcoded tables.
// ---------------------------------------------------------------------------
__global__ void __launch_bounds__(128)
ms_deform_attn(const float* __restrict__ offs, const float* __restrict__ awl,
               const float* __restrict__ ref, const float* __restrict__ value,
               float* __restrict__ out, int nPts)
{
  constexpr int LH[4] = {100, 50, 25, 13};
  constexpr int LW[4] = {150, 75, 38, 19};
  constexpr int LS[4] = {0, 15000, 18750, 19700};

  const int wave = threadIdx.x >> 5, lane = threadIdx.x & 31;
  const int h = blockIdx.y, b = blockIdx.z;
  const int Lq = N_OBJ * nPts;
  const int lq = blockIdx.x * 4 + wave;
  if (lq >= Lq) return;
  const int o = lq / nPts;

  // softmax over the 16 (level,point) logits for this head
  float logit = (lane < 16) ? awl[(size_t)(b * Lq + lq) * 128 + h * 16 + lane] : -1e30f;
  float mx = logit;
#pragma unroll
  for (int d = 16; d; d >>= 1) mx = fmaxf(mx, __shfl_xor(mx, d, 32));
  float e = (lane < 16) ? __expf(logit - mx) : 0.f;
  float sum = e;
#pragma unroll
  for (int d = 16; d; d >>= 1) sum += __shfl_xor(sum, d, 32);
  float p = e / sum;

  // 32 offset floats for this head live exactly one-per-lane
  float ov = offs[(size_t)(b * Lq + lq) * 256 + h * 32 + lane];

  float acc = 0.f;
#pragma unroll
  for (int i = 0; i < 16; ++i) {
    int l = i >> 2;
    float w  = __shfl(p, i, 32);
    float ox = __shfl(ov, 2 * i, 32);
    float oy = __shfl(ov, 2 * i + 1, 32);
    float rx = ref[((size_t)(b * N_OBJ + o) * 4 + l) * 2 + 0];
    float ry = ref[((size_t)(b * N_OBJ + o) * 4 + l) * 2 + 1];
    int Wl = LW[l], Hl = LH[l], st = LS[l];
    float x = (rx + ox / (float)Wl) * (float)Wl - 0.5f;
    float y = (ry + oy / (float)Hl) * (float)Hl - 0.5f;
    float x0f = floorf(x), y0f = floorf(y);
    float fx = x - x0f, fy = y - y0f;
    int x0 = (int)x0f, y0 = (int)y0f;
    float bl = 0.f;
#pragma unroll
    for (int c = 0; c < 4; ++c) {
      int yi = y0 + (c >> 1), xi = x0 + (c & 1);
      if (yi < 0 || yi >= Hl || xi < 0 || xi >= Wl) continue;
      float wc = ((c & 1) ? fx : 1.f - fx) * ((c >> 1) ? fy : 1.f - fy);
      bl += wc * value[((size_t)b * S_TOT + st + yi * Wl + xi) * 256 + h * 32 + lane];
    }
    acc += w * bl;
  }
  out[(size_t)(b * Lq + lq) * 256 + h * 32 + lane] = acc;
}

// ---------------------------------------------------------------------------
// Host orchestration
// ---------------------------------------------------------------------------
struct BranchP {
  const float *cWa, *cWo, *cWout, *cWv, *cba, *cbo, *cbout, *cbv;     // cross
  const float *fW1, *fW2, *fb1, *fb2, *fbn, *fg;                      // ffn
  const float *ieWout, *ieWqkv, *iebout, *iebqkv;                     // inter
  const float *iaWout, *iaWqkv, *iabout, *iabqkv;                     // intra
  const float *lnc_g, *lnc_b, *lni_g, *lni_b, *lna_g, *lna_b;
};

static inline void gemm(const float* A, const float* W, const float* bias,
                        float* C, int M, int N, int K, bool relu, hipStream_t s)
{
  dim3 g((M + 127) / 128, N / 64);
  if (relu) gemm_bf16_wmma<1><<<g, dim3(256), 0, s>>>(A, W, bias, C, M, N, K);
  else      gemm_bf16_wmma<0><<<g, dim3(256), 0, s>>>(A, W, bias, C, M, N, K);
}

static void run_branch(const float* tgt, const float* qpos, const float* ref,
                       const float* src, int nPts, const BranchP& P,
                       float* out, float* ws, hipStream_t st)
{
  const int T = BATCH * N_OBJ * nPts;
  const int n = T * D_MODEL;
  const size_t TOK = (size_t)BATCH * N_OBJ * N_TEXT * D_MODEL;   // max token buf

  float* value = ws;                                      // B*S*256 floats
  float* xq    = value + (size_t)BATCH * S_TOT * D_MODEL;
  float* q     = xq  + TOK;
  float* k     = q   + TOK;
  float* v     = k   + TOK;
  float* ao    = v   + TOK;
  float* t2    = ao  + TOK;
  float* x1    = t2  + TOK;
  float* x2    = x1  + TOK;
  float* awl   = x2  + TOK;                               // T*128 floats
  float* h1    = value;   // FFN hidden aliases value (value dead by then)

  const int n4  = n / 4;
  const int nb4 = (n4 + 255) / 256;
  const int lnb = (T + 3) / 4;

  // ---- intra self-attention (across points within each object) ----
  add_vec4<<<nb4, 256, 0, st>>>((const float4*)tgt, (const float4*)qpos, (float4*)xq, n4);
  gemm(xq,  P.iaWqkv,             P.iabqkv,       q, T, 256, 256, false, st);
  gemm(xq,  P.iaWqkv + 256 * 256, P.iabqkv + 256, k, T, 256, 256, false, st);
  gemm(tgt, P.iaWqkv + 512 * 256, P.iabqkv + 512, v, T, 256, 256, false, st);
  flash_attn_small<<<dim3((nPts + 3) / 4, N_HEADS, BATCH * N_OBJ), 128, 0, st>>>(q, k, v, ao, 0, N_OBJ, nPts);
  gemm(ao, P.iaWout, P.iabout, t2, T, 256, 256, false, st);
  ln_residual<<<lnb, 128, 0, st>>>(t2, tgt, P.lna_g, P.lna_b, x1, T);

  // ---- inter self-attention (across objects per point slot) ----
  gemm(x1, P.ieWqkv,             P.iebqkv,       q, T, 256, 256, false, st);
  gemm(x1, P.ieWqkv + 256 * 256, P.iebqkv + 256, k, T, 256, 256, false, st);
  gemm(x1, P.ieWqkv + 512 * 256, P.iebqkv + 512, v, T, 256, 256, false, st);
  flash_attn_small<<<dim3((N_OBJ + 3) / 4, N_HEADS, BATCH * nPts), 128, 0, st>>>(q, k, v, ao, 1, N_OBJ, nPts);
  gemm(ao, P.ieWout, P.iebout, t2, T, 256, 256, false, st);
  ln_residual<<<lnb, 128, 0, st>>>(t2, x1, P.lni_g, P.lni_b, x2, T);

  // ---- deformable cross-attention ----
  add_vec4<<<nb4, 256, 0, st>>>((const float4*)x2, (const float4*)qpos, (float4*)xq, n4);
  gemm(src, P.cWv, P.cbv, value, BATCH * S_TOT, 256, 256, false, st);  // value proj (the big GEMM)
  gemm(xq,  P.cWo, P.cbo, q,   T, 256, 256, false, st);                // sampling offsets
  gemm(xq,  P.cWa, P.cba, awl, T, 128, 256, false, st);                // attention logits
  ms_deform_attn<<<dim3((N_OBJ * nPts + 3) / 4, N_HEADS, BATCH), 128, 0, st>>>(q, awl, ref, value, ao, nPts);
  gemm(ao, P.cWout, P.cbout, t2, T, 256, 256, false, st);
  ln_residual<<<lnb, 128, 0, st>>>(t2, x2, P.lnc_g, P.lnc_b, x1, T);   // x1 = x3

  // ---- FFN ----
  gemm(x1, P.fW1, P.fb1, h1, T, D_FFN, 256, true,  st);
  gemm(h1, P.fW2, P.fb2, t2, T, 256, D_FFN, false, st);
  ln_residual<<<lnb, 128, 0, st>>>(t2, x1, P.fg, P.fbn, out, T);
}

extern "C" void kernel_launch(void* const* d_in, const int* in_sizes, int n_in,
                              void* d_out, int out_size, void* d_ws, size_t ws_size,
                              hipStream_t stream)
{
  (void)in_sizes; (void)n_in; (void)out_size; (void)ws_size;
  const float* tgt   = (const float*)d_in[0];
  const float* qpos  = (const float*)d_in[1];
  const float* tgtT  = (const float*)d_in[2];
  const float* qposT = (const float*)d_in[3];
  const float* ref   = (const float*)d_in[4];
  const float* src   = (const float*)d_in[5];
  // d_in[6], d_in[7]: spatial shapes / level starts (static -> hardcoded)

  // params pytree leaves: sorted dict keys, 'text' branch first, then 'vis'.
  auto fill = [&](int b) {
    BranchP P;
    P.cWa   = (const float*)d_in[b + 0];  P.cWo    = (const float*)d_in[b + 1];
    P.cWout = (const float*)d_in[b + 2];  P.cWv    = (const float*)d_in[b + 3];
    P.cba   = (const float*)d_in[b + 4];  P.cbo    = (const float*)d_in[b + 5];
    P.cbout = (const float*)d_in[b + 6];  P.cbv    = (const float*)d_in[b + 7];
    P.fW1   = (const float*)d_in[b + 8];  P.fW2    = (const float*)d_in[b + 9];
    P.fb1   = (const float*)d_in[b + 10]; P.fb2    = (const float*)d_in[b + 11];
    P.fbn   = (const float*)d_in[b + 12]; P.fg     = (const float*)d_in[b + 13];
    P.ieWout= (const float*)d_in[b + 14]; P.ieWqkv = (const float*)d_in[b + 15];
    P.iebout= (const float*)d_in[b + 16]; P.iebqkv = (const float*)d_in[b + 17];
    P.iaWout= (const float*)d_in[b + 18]; P.iaWqkv = (const float*)d_in[b + 19];
    P.iabout= (const float*)d_in[b + 20]; P.iabqkv = (const float*)d_in[b + 21];
    P.lnc_g = (const float*)d_in[b + 22]; P.lnc_b  = (const float*)d_in[b + 23];
    P.lni_g = (const float*)d_in[b + 24]; P.lni_b  = (const float*)d_in[b + 25];
    P.lna_g = (const float*)d_in[b + 26]; P.lna_b  = (const float*)d_in[b + 27];
    return P;
  };
  BranchP Ptext = fill(8);
  BranchP Pvis  = fill(8 + 28);

  float* ws   = (float*)d_ws;
  float* outv = (float*)d_out;
  float* outt = outv + (size_t)BATCH * N_OBJ * N_PTS * D_MODEL;

  run_branch(tgt,  qpos,  ref, src, N_PTS,  Pvis,  outv, ws, stream);
  run_branch(tgtT, qposT, ref, src, N_TEXT, Ptext, outt, ws, stream);
}
